// TrainingGraphModel_23733989277702
// MI455X (gfx1250) — compile-verified, matrix operand order checked by reading, not run on hardware
//
#include <hip/hip_runtime.h>
#include <hip/hip_bf16.h>
#include <math.h>

// MI455X / gfx1250, wave32.  G=256, M=N=D=256 fixed by the reference.
// Single fused kernel: norms -> WMMA cosine-corr -> log_p in LDS ->
// 8 Sinkhorn iterations in LDS -> exp()*512 to global.

typedef __attribute__((ext_vector_type(16))) _Float16 v16h;
typedef __attribute__((ext_vector_type(8)))  float    v8f;

#define DIMD     256
#define AUG      257                      // M+1 = N+1
#define GSTRIDE  (AUG*AUG)                // 66049 floats per group
#define LOG_NORM  (-6.238324625039508f)   // log(1/512)
#define LOG_HALF  (-0.6931471805599453f)  // log(256/512)
#define NTHREADS 512                      // 16 wave32s

__global__ __launch_bounds__(NTHREADS)
void sinkhorn_fused_kernel(const float* __restrict__ tra,
                           const float* __restrict__ det,
                           const float* __restrict__ alpha,
                           const float* __restrict__ eplison,
                           float* __restrict__ out) {
    extern __shared__ float lp[];             // 66049 floats = 264196 B dynamic LDS
    __shared__ float n1s[DIMD], n2s[DIMD];    // 2 KB static LDS

    const int g    = blockIdx.x;
    const int tid  = threadIdx.x;
    const int lane = tid & 31;
    const int wave = tid >> 5;                // 0..15

    const float* A = tra + ((size_t)g << 16);
    const float* B = det + ((size_t)g << 16);

    // ---------------- phase 0: row norms (wave per row, shuffle reduce) ----
    for (int row = wave; row < 2 * DIMD; row += 16) {
        const float* p = (row < DIMD) ? (A + (size_t)row * DIMD)
                                      : (B + (size_t)(row - DIMD) * DIMD);
        float s = 0.f;
        #pragma unroll
        for (int j = 0; j < DIMD; j += 32) {
            const float v = p[j + lane];
            s += v * v;
        }
        #pragma unroll
        for (int off = 16; off > 0; off >>= 1)
            s += __shfl_xor(s, off, 32);
        if (lane == 0) {
            if (row < DIMD) n1s[row] = sqrtf(s);
            else            n2s[row - DIMD] = sqrtf(s);
        }
    }
    __syncthreads();

    const float inv_l = 1.0f / (__expf(eplison[0]) + 0.03f);

    // ---------------- phase 1: WMMA cosine correlation -> log_p in LDS -----
    {
        const int h   = lane >> 4;            // ISA 16-bit A-layout lane group
        const int l15 = lane & 15;
        const int Mbase = wave * 16;          // tile_m == wave (16 waves, 16 tile rows)

        // Hoist all A fragments for this tile row: 8 x v16h = 64 VGPRs.
        // Lanes 0-15 hold K=0..7 (V0-3) and K=16..23 (V4-7) of each 32-chunk;
        // lanes 16-31 hold K=8..15 and K=24..31 (per ISA 16-bit 16x32 table).
        const float* Arow = A + (size_t)(Mbase + l15) * DIMD;
        v16h afr[8];
        #pragma unroll
        for (int kk = 0; kk < 8; ++kk) {
            const int kb = kk * 32;
            #pragma unroll
            for (int j = 0; j < 8; ++j) {
                afr[kk][j]     = (_Float16)Arow[kb + 8 * h + j];
                afr[kk][j + 8] = (_Float16)Arow[kb + 16 + 8 * h + j];
            }
        }

        for (int tn = 0; tn < 16; ++tn) {
            const int N = tn * 16 + l15;
            const float* Brow = B + (size_t)N * DIMD;
            v8f c = {};
            #pragma unroll
            for (int kk = 0; kk < 8; ++kk) {
                const int kb = kk * 32;
                v16h bfr;
                #pragma unroll
                for (int j = 0; j < 8; ++j) {
                    bfr[j]     = (_Float16)Brow[kb + 8 * h + j];
                    bfr[j + 8] = (_Float16)Brow[kb + 16 + 8 * h + j];
                }
                c = __builtin_amdgcn_wmma_f32_16x16x32_f16(
                        false, afr[kk], false, bfr, (short)0, c, false, false);
            }
            // C/D layout: VGPR r -> row M = Mbase + r + 8h, col N (lanes&15).
            const float rn2 = 1.0f / n2s[N];
            #pragma unroll
            for (int r = 0; r < 8; ++r) {
                const int Mr = Mbase + r + 8 * h;
                const float corr = c[r] * rn2 / n1s[Mr];
                lp[Mr * AUG + N] = (corr - 1.0f) * inv_l;   // -(1-corr)/lambda
            }
        }
    }

    // Dustbin row/col:  -(1-alpha)/lambda
    {
        const float dv = (alpha[0] - 1.0f) * inv_l;
        for (int i = tid; i < AUG + DIMD; i += NTHREADS) {
            if (i < AUG) lp[256 * AUG + i] = dv;            // last row (all 257)
            else         lp[(i - AUG) * AUG + 256] = dv;    // last col (rows 0..255)
        }
    }
    __syncthreads();

    // ---------------- phase 2: 8 Sinkhorn iterations, fully in LDS ---------
    for (int it = 0; it < 8; ++it) {
        // axis-1 (over m) : thread per column; stride-257 => bank (m+n)%64,
        // conflict-free across consecutive threads.
        if (tid < AUG) {
            const int n = tid;
            float mx = -INFINITY, s = 0.f;
            for (int m = 0; m < AUG; ++m) {                 // streaming logsumexp
                const float x  = lp[m * AUG + n];
                const float mn = fmaxf(mx, x);
                s = s * __expf(mx - mn) + __expf(x - mn);
                mx = mn;
            }
            const float adj = (n == 256 ? LOG_HALF : LOG_NORM) - (mx + __logf(s));
            for (int m = 0; m < AUG; ++m)
                lp[m * AUG + n] += adj;
        }
        __syncthreads();

        // axis-2 (over n) : wave per row, butterfly-merged streaming logsumexp.
        for (int m = wave; m < AUG; m += 16) {
            float mx = -INFINITY, s = 0.f;
            for (int n = lane; n < AUG; n += 32) {
                const float x  = lp[m * AUG + n];
                const float mn = fmaxf(mx, x);
                s = s * __expf(mx - mn) + __expf(x - mn);
                mx = mn;
            }
            #pragma unroll
            for (int off = 16; off > 0; off >>= 1) {        // merge (max, scaled-sum)
                const float mo = __shfl_xor(mx, off, 32);
                const float so = __shfl_xor(s,  off, 32);
                const float mn = fmaxf(mx, mo);
                s = s * __expf(mx - mn) + so * __expf(mo - mn);
                mx = mn;
            }
            const float adj = (m == 256 ? LOG_HALF : LOG_NORM) - (mx + __logf(s));
            for (int n = lane; n < AUG; n += 32)
                lp[m * AUG + n] += adj;
        }
        __syncthreads();
    }

    // ---------------- phase 3: exp(lp)*(M+N) -> global, coalesced ----------
    float* og = out + (size_t)g * GSTRIDE;
    for (int i = tid; i < GSTRIDE; i += NTHREADS)
        og[i] = __expf(lp[i]) * 512.0f;
}

// -------------------------------------------------------------- launch -----
extern "C" void kernel_launch(void* const* d_in, const int* in_sizes, int n_in,
                              void* d_out, int out_size, void* d_ws, size_t ws_size,
                              hipStream_t stream) {
    const float* tra     = (const float*)d_in[0];
    const float* det     = (const float*)d_in[1];
    const float* alpha   = (const float*)d_in[2];
    const float* eplison = (const float*)d_in[3];
    float* out = (float*)d_out;

    const int G = in_sizes[0] / (DIMD * DIMD);      // 256

    // One workgroup per group g; 264 KB dynamic LDS (<= 320 KB/WGP on CDNA5).
    const size_t lds_bytes = (size_t)GSTRIDE * sizeof(float);
    sinkhorn_fused_kernel<<<G, NTHREADS, lds_bytes, stream>>>(
        tra, det, alpha, eplison, out);
}
